// BigFaceEncoderDecoder_13099650253571
// MI455X (gfx1250) — matrix operand order repaired and checked by reading, not compiled, not run on hardware
//
#include <hip/hip_runtime.h>
#include <stdint.h>
#include <stddef.h>

// ---------------------------------------------------------------------------
// Types for CDNA5 WMMA (wave32): A/B are 16 bf16 per lane, C/D are 8 f32.
// ---------------------------------------------------------------------------
typedef __attribute__((ext_vector_type(16))) __bf16        v16bf;
typedef __attribute__((ext_vector_type(8)))  float         v8f;
typedef __attribute__((ext_vector_type(4)))  unsigned int  v4u;
typedef __attribute__((ext_vector_type(2)))  unsigned int  v2u;

#define THREADS 256
#define ROWS    64   // rows staged per workgroup

__device__ __forceinline__ unsigned short f2bf(float f) {
  // round-to-nearest-even f32 -> bf16
  unsigned int u = __builtin_bit_cast(unsigned int, f);
  u += 0x7fffu + ((u >> 16) & 1u);
  return (unsigned short)(u >> 16);
}

__device__ __forceinline__ float leaky(float v) {
  return (v >= 0.f) ? v : 0.01f * v;
}

// ---------------------------------------------------------------------------
// Pack conv weights w[cout][cin][9] (f32) into bf16 B-matrix, layout
// packed[co * Kpad + kf] with kf = neigh*cin + ci (K contiguous per column),
// so a WMMA lane's 16-element B fragment is two 16-byte contiguous loads.
// ---------------------------------------------------------------------------
__global__ void pack_w_kernel(const float* __restrict__ w,
                              unsigned short* __restrict__ p,
                              int cin, int cout, int Kpad) {
  int i = blockIdx.x * THREADS + threadIdx.x;
  int total = cout * Kpad;
  if (i >= total) return;
  int co = i / Kpad;
  int kf = i - co * Kpad;
  unsigned short v = 0;
  int K9 = cin * 9;
  if (kf < K9) {
    int neigh = kf / cin;
    int ci = kf - neigh * cin;
    v = f2bf(w[(co * cin + ci) * 9 + neigh]);
  }
  p[i] = v;
}

// ---------------------------------------------------------------------------
// Deterministic BN statistics (two stage, no float atomics).
// Stage 1: per 256-row block, per-channel sum / sumsq. C in {64,128,256}.
// ---------------------------------------------------------------------------
__global__ void bn_partial_kernel(const float* __restrict__ act, int N, int C,
                                  float* __restrict__ part) {
  __shared__ float ls[THREADS];
  __shared__ float lq[THREADS];
  int tid = threadIdx.x;
  int c = tid & (C - 1);
  int sub = tid / C;
  int S = THREADS / C;
  int base = blockIdx.x * THREADS;
  float s = 0.f, q = 0.f;
  for (int r = sub; r < THREADS; r += S) {
    int row = base + r;
    if (row < N) {
      float v = act[(size_t)row * C + c];
      s += v;
      q += v * v;
    }
  }
  ls[tid] = s;
  lq[tid] = q;
  __syncthreads();
  if (sub == 0) {
    for (int s2 = 1; s2 < S; ++s2) { s += ls[s2 * C + c]; q += lq[s2 * C + c]; }
    part[(size_t)blockIdx.x * 2 * C + c] = s;
    part[(size_t)blockIdx.x * 2 * C + C + c] = q;
  }
}

// Stage 2: fold partials (fixed order), emit fused scale/shift:
//   y = x*scale + shift  ==  (x-mean)*rsqrt(var+eps)*g + b
__global__ void bn_finalize_kernel(const float* __restrict__ part, int nblk,
                                   int C, int N,
                                   const float* __restrict__ g,
                                   const float* __restrict__ bb,
                                   float* __restrict__ scale,
                                   float* __restrict__ shift) {
  int c = threadIdx.x;
  if (c >= C) return;
  float s = 0.f, q = 0.f;
  for (int b = 0; b < nblk; ++b) {
    s += part[(size_t)b * 2 * C + c];
    q += part[(size_t)b * 2 * C + C + c];
  }
  float inv = 1.0f / (float)N;
  float m = s * inv;
  float var = q * inv - m * m;
  float sc = g[c] * rsqrtf(var + 1e-5f);
  scale[c] = sc;
  shift[c] = bb[c] - m * sc;
}

// ---------------------------------------------------------------------------
// Fused face conv: [optional BN+LeakyReLU] -> 9-neighbor gather -> bf16 GEMM
// via v_wmma_f32_16x16x32_bf16 -> bias [+ residual] -> f32 store.
// One workgroup (8 wave32) stages 64 rows; each wave item computes a pair of
// 16x16 output tiles (two independent accumulators sharing one B fragment).
// ---------------------------------------------------------------------------
template <bool PRELUDE>
__global__ __launch_bounds__(THREADS)
void fconv_kernel(const float* __restrict__ act, const int* __restrict__ nbh,
                  const unsigned short* __restrict__ packB,
                  const float* __restrict__ bias,
                  const float* __restrict__ scale,
                  const float* __restrict__ shift,
                  const float* __restrict__ residual,
                  float* __restrict__ out,
                  int N, int Cin, int Cout, int Kpad) {
  extern __shared__ unsigned char smem_raw[];
  unsigned short* As = (unsigned short*)smem_raw;  // [ROWS][Kpad] bf16

  const int tid = threadIdx.x;
  const int wave = tid >> 5;
  const int lane = tid & 31;
  const int rowbase = blockIdx.x * ROWS;

  // ---- Stage gathered+activated A tile into LDS (bf16) ----
  if ((Cin & 31) == 0) {
    // Structured path: Cin is a power of two (64/128/256). Each lane handles a
    // float4 channel group: one global_load_b128, vectorized BN+LeakyReLU,
    // pack to 2 dwords, one ds_store_b64. neigh/ci via shift/mask (no divides).
    const int ngroups = (Cin * 9) >> 2;      // float4 groups per row
    const int lcin = 31 - __builtin_clz((unsigned)Cin);
    const int cmask = Cin - 1;
    for (int r = wave; r < ROWS; r += 8) {
      int grow = rowbase + r;
      unsigned short* rowp = As + (size_t)r * Kpad;
      if (grow < N) {
        for (int kq = lane; kq < ngroups; kq += 32) {
          int kf = kq << 2;
          int neigh = kf >> lcin;
          int ci = kf & cmask;
          int nrow = nbh[grow * 9 + neigh];
          unsigned int d0 = 0, d1 = 0;
          if (nrow < N) {  // rows >= N are zero padding
            float4 v = *(const float4*)(act + (size_t)nrow * Cin + ci);
            float vx = v.x, vy = v.y, vz = v.z, vw = v.w;
            if (PRELUDE) {  // fused BN + LeakyReLU(0.01)
              float4 sc = *(const float4*)(scale + ci);
              float4 sh = *(const float4*)(shift + ci);
              vx = leaky(vx * sc.x + sh.x);
              vy = leaky(vy * sc.y + sh.y);
              vz = leaky(vz * sc.z + sh.z);
              vw = leaky(vw * sc.w + sh.w);
            }
            d0 = (unsigned int)f2bf(vx) | ((unsigned int)f2bf(vy) << 16);
            d1 = (unsigned int)f2bf(vz) | ((unsigned int)f2bf(vw) << 16);
          }
          v2u dd = {d0, d1};
          *(v2u*)(rowp + kf) = dd;
        }
      } else {
        for (int kq = lane; kq < (Kpad >> 2); kq += 32) {
          v2u dd = {0u, 0u};
          *(v2u*)(rowp + (kq << 2)) = dd;
        }
      }
    }
  } else {
    // Generic path (enc_conv_in: Cin == 3, K9 = 27, Kpad = 32).
    const int K9 = Cin * 9;
    const int total = ROWS * Kpad;
    for (int i = tid; i < total; i += THREADS) {
      int r = i / Kpad;
      int kf = i - r * Kpad;
      unsigned short val = 0;
      int grow = rowbase + r;
      if (kf < K9 && grow < N) {
        int neigh = kf / Cin;
        int ci = kf - neigh * Cin;
        int nrow = nbh[grow * 9 + neigh];
        if (nrow < N) {
          float v = act[(size_t)nrow * Cin + ci];
          if (PRELUDE) v = leaky(v * scale[ci] + shift[ci]);
          val = f2bf(v);
        }
      }
      As[i] = val;
    }
  }
  __syncthreads();

  // ---- WMMA phase ----
  const int half = lane >> 4;   // lane 16..31 -> upper half
  const int l15 = lane & 15;
  const int nct = Cout >> 4;    // 16-wide column tiles (Cout in {64,128,256})
  const int items = nct << 1;   // 2 row-tile-pairs per block

  union U16 { v4u q[2]; v16bf v; };

  for (int item = wave; item < items; item += 8) {
    const int rp = item & 1;    // row-tile pair: rows rp*32 .. rp*32+31
    const int ct = item >> 1;
    v8f acc0 = {0.f, 0.f, 0.f, 0.f, 0.f, 0.f, 0.f, 0.f};
    v8f acc1 = {0.f, 0.f, 0.f, 0.f, 0.f, 0.f, 0.f, 0.f};
    // A layout (16-bit 16x32): lane m=l15; VGPR0-3 hold K = 8*half+0..7,
    // VGPR4-7 hold K = 16+8*half+0..7  -> two contiguous b128 loads.
    const unsigned short* arow0 = As + (size_t)((rp << 5) + l15) * Kpad;
    const unsigned short* arow1 = As + (size_t)((rp << 5) + 16 + l15) * Kpad;
    // B layout (16-bit 32x16): lane n=l15; K = 8*half+0..7, 16+8*half+0..7.
    const unsigned short* brow =
        packB + (size_t)((ct << 4) + l15) * Kpad + (half << 3);

    for (int kt = 0; kt < Kpad; kt += 32) {
      U16 ua0, ua1, ub;
      ub.q[0] = *(const v4u*)(brow + kt);
      ub.q[1] = *(const v4u*)(brow + kt + 16);
      ua0.q[0] = *(const v4u*)(arow0 + kt + (half << 3));
      ua0.q[1] = *(const v4u*)(arow0 + kt + 16 + (half << 3));
      ua1.q[0] = *(const v4u*)(arow1 + kt + (half << 3));
      ua1.q[1] = *(const v4u*)(arow1 + kt + 16 + (half << 3));
      // Two independent accumulation chains sharing the B fragment.
      acc0 = __builtin_amdgcn_wmma_f32_16x16x32_bf16(
          false, ua0.v, false, ub.v, (short)0, acc0, false, false);
      acc1 = __builtin_amdgcn_wmma_f32_16x16x32_bf16(
          false, ua1.v, false, ub.v, (short)0, acc1, false, false);
    }

    // D layout: col = l15 (+tile), row = vgpr + 8*half (+tile).
    const int co = (ct << 4) + l15;
    const float bv = bias[co];
#pragma unroll
    for (int r = 0; r < 8; ++r) {
      int grow = rowbase + (rp << 5) + (half << 3) + r;
      if (grow < N) {
        size_t oi = (size_t)grow * Cout + co;
        float o = acc0[r] + bv;
        if (residual) o += residual[oi];
        out[oi] = o;
      }
    }
#pragma unroll
    for (int r = 0; r < 8; ++r) {
      int grow = rowbase + (rp << 5) + 16 + (half << 3) + r;
      if (grow < N) {
        size_t oi = (size_t)grow * Cout + co;
        float o = acc1[r] + bv;
        if (residual) o += residual[oi];
        out[oi] = o;
      }
    }
  }
}

// ---------------------------------------------------------------------------
// 4:1 max pool over contiguous row groups (pool_maps = repeat(arange,4)).
// ---------------------------------------------------------------------------
__global__ void pool4_kernel(const float* __restrict__ in, float* __restrict__ out,
                             int Nout, int C) {
  int i = blockIdx.x * THREADS + threadIdx.x;
  if (i >= Nout * C) return;
  int j = i / C;
  int c = i - j * C;
  size_t b = (size_t)(j * 4) * C + c;
  float m = in[b];
  m = fmaxf(m, in[b + C]);
  m = fmaxf(m, in[b + 2 * (size_t)C]);
  m = fmaxf(m, in[b + 3 * (size_t)C]);
  out[i] = m;
}

// ---------------------------------------------------------------------------
// Fused unpool (x4) + pad-corrected blur at the fine level.
// ---------------------------------------------------------------------------
__global__ void blur_unpool_kernel(const float* __restrict__ coarse,
                                   const int* __restrict__ nbh,
                                   float* __restrict__ out, int Nf, int C) {
  const float BF[9] = {0.25f, 0.125f, 0.0625f, 0.125f, 0.0625f,
                       0.125f, 0.0625f, 0.125f, 0.0625f};
  int i = blockIdx.x * THREADS + threadIdx.x;
  if (i >= Nf * C) return;
  int n = i / C;
  int c = i - n * C;
  float ws = 0.f, corr = 0.f;
#pragma unroll
  for (int k = 0; k < 9; ++k) {
    int idx = nbh[n * 9 + k];
    float wgt = BF[k];
    if (idx < Nf) {  // non-pad
      ws += wgt * coarse[(size_t)(idx >> 2) * C + c];
      corr += wgt;
    }
  }
  out[i] = ws / corr;
}

// ---------------------------------------------------------------------------
// Head: BN+LeakyReLU (precomputed scale/shift) -> 64x3 linear -> tanh.
// ---------------------------------------------------------------------------
__global__ void head_kernel(const float* __restrict__ h,
                            const float* __restrict__ scale,
                            const float* __restrict__ shift,
                            const float* __restrict__ W,
                            const float* __restrict__ b,
                            float* __restrict__ out, int N) {
  int n = blockIdx.x * THREADS + threadIdx.x;
  if (n >= N) return;
  float o0 = b[0], o1 = b[1], o2 = b[2];
  const float* row = h + (size_t)n * 64;
#pragma unroll 4
  for (int c4 = 0; c4 < 16; ++c4) {
    float4 v = *(const float4*)(row + c4 * 4);
    float4 sc = *(const float4*)(scale + c4 * 4);
    float4 sh = *(const float4*)(shift + c4 * 4);
    float4 w0 = *(const float4*)(W + c4 * 4);
    float4 w1 = *(const float4*)(W + 64 + c4 * 4);
    float4 w2 = *(const float4*)(W + 128 + c4 * 4);
    float a0 = leaky(v.x * sc.x + sh.x);
    float a1 = leaky(v.y * sc.y + sh.y);
    float a2 = leaky(v.z * sc.z + sh.z);
    float a3 = leaky(v.w * sc.w + sh.w);
    o0 += a0 * w0.x + a1 * w0.y + a2 * w0.z + a3 * w0.w;
    o1 += a0 * w1.x + a1 * w1.y + a2 * w1.z + a3 * w1.w;
    o2 += a0 * w2.x + a1 * w2.y + a2 * w2.z + a3 * w2.w;
  }
  out[(size_t)n * 3 + 0] = tanhf(o0);
  out[(size_t)n * 3 + 1] = tanhf(o1);
  out[(size_t)n * 3 + 2] = tanhf(o2);
}

// ---------------------------------------------------------------------------
// Host orchestration.
// ---------------------------------------------------------------------------
extern "C" void kernel_launch(void* const* d_in, const int* in_sizes, int n_in,
                              void* d_out, int out_size, void* d_ws, size_t ws_size,
                              hipStream_t stream) {
  (void)out_size;
  static const int NL[6] = {49152, 12288, 3072, 768, 192, 48};
  struct BI { int cin, cout; };
  static const BI binfo[30] = {
      {64, 64},   {64, 64},                 // down_0
      {64, 64},   {64, 64},                 // down_1
      {64, 128},  {128, 128},               // down_2
      {128, 128}, {128, 128},               // down_3
      {128, 256}, {256, 256},               // down_4
      {256, 256}, {256, 256},               // enc_mid
      {256, 256}, {256, 256},               // dec_mid
      {256, 256}, {256, 256}, {256, 256},   // up_4
      {256, 128}, {128, 128}, {128, 128},   // up_3
      {128, 128}, {128, 128}, {128, 128},   // up_2
      {128, 64},  {64, 64},   {64, 64},     // up_1
      {64, 64},   {64, 64},   {64, 64},     // up_0
      {64, 64}                              // dec_out_block
  };

  // ---- Parse flattened inputs (setup_inputs insertion order) ----
  int idx = 0;
  auto F = [&](int k) { return (const float*)d_in[k]; };
  const float* x = F(idx++);
  const int* nbh[6];
  for (int l = 0; l < 6; ++l) nbh[l] = (const int*)d_in[idx++];
  // pool_maps (deterministic repeat(arange,4) — values unused)
  {
    const int pm[5] = {49152, 12288, 3072, 768, 192};
    for (int l = 0; l < 5; ++l)
      if (idx < n_in && in_sizes[idx] == pm[l]) idx++;
  }
  // is_pad (reconstructible as idx >= N — unused)
  {
    const int ps[6] = {55296, 13824, 3456, 864, 216, 56};
    for (int l = 0; l < 6; ++l)
      if (idx < n_in && in_sizes[idx] == ps[l]) idx++;
  }
  // pads (python ints): 6 scalars or one array of 6 (or absent)
  if (idx < n_in && in_sizes[idx] == 1) idx += 6;
  else if (idx < n_in && in_sizes[idx] == 6) idx += 1;
  // node_counts: 5 scalars or one array of 5 (or absent)
  if (idx < n_in && in_sizes[idx] == 1) idx += 5;
  else if (idx < n_in && in_sizes[idx] == 5) idx += 1;

  struct ConvP { const float* w; const float* b; int cin, cout, Kpad;
                 unsigned short* pack; };
  struct BnP { const float* g; const float* b; };
  ConvP convs[80];
  int nc = 0;
  auto addConv = [&](int cin, int cout) -> int {
    ConvP c;
    c.w = F(idx++); c.b = F(idx++);
    c.cin = cin; c.cout = cout;
    c.Kpad = ((cin * 9 + 31) / 32) * 32;
    c.pack = nullptr;
    convs[nc] = c;
    return nc++;
  };

  struct BlockP { BnP n0, n1; int c0, c1, ns; };
  BlockP blk[30];
  for (int b = 0; b < 30; ++b) {
    int cin = binfo[b].cin, cout = binfo[b].cout;
    blk[b].n0.g = F(idx++); blk[b].n0.b = F(idx++);
    blk[b].c0 = addConv(cin, cout);
    blk[b].n1.g = F(idx++); blk[b].n1.b = F(idx++);
    blk[b].c1 = addConv(cout, cout);
    blk[b].ns = (cin != cout) ? addConv(cin, cout) : -1;
  }
  int encConvIn = addConv(3, 64);
  BnP encOutNorm; encOutNorm.g = F(idx++); encOutNorm.b = F(idx++);
  int encOutConv = addConv(256, 128);
  int decConvIn = addConv(128, 256);
  BnP decOutNorm; decOutNorm.g = F(idx++); decOutNorm.b = F(idx++);
  const float* linW = F(idx++);
  const float* linB = F(idx++);

  // ---- Workspace carve (bump allocator) ----
  char* base = (char*)d_ws;
  size_t off = 0;
  auto alloc = [&](size_t bytes) -> void* {
    void* p = base + off;
    off = (off + bytes + 255) & ~(size_t)255;
    return p;
  };
  for (int i = 0; i < nc; ++i)
    convs[i].pack =
        (unsigned short*)alloc((size_t)convs[i].cout * convs[i].Kpad * 2);
  const size_t BUFE = (size_t)49152 * 64;  // largest N*C activation
  float* buf[3];
  for (int i = 0; i < 3; ++i) buf[i] = (float*)alloc(BUFE * sizeof(float));
  float* part = (float*)alloc((size_t)192 * 512 * sizeof(float));
  float* scaleB = (float*)alloc(256 * sizeof(float));
  float* shiftB = (float*)alloc(256 * sizeof(float));
  if (off > ws_size) return;  // workspace too small: no-op

  // ---- Pack all conv weights to bf16 WMMA-B layout (deterministic) ----
  for (int i = 0; i < nc; ++i) {
    int total = convs[i].cout * convs[i].Kpad;
    pack_w_kernel<<<(total + THREADS - 1) / THREADS, THREADS, 0, stream>>>(
        convs[i].w, convs[i].pack, convs[i].cin, convs[i].cout, convs[i].Kpad);
  }

  // ---- Helpers ----
  auto runBn = [&](const float* act, int N, int C, const BnP& p) {
    int nblk = (N + THREADS - 1) / THREADS;
    bn_partial_kernel<<<nblk, THREADS, 0, stream>>>(act, N, C, part);
    bn_finalize_kernel<<<1, THREADS, 0, stream>>>(part, nblk, C, N, p.g, p.b,
                                                  scaleB, shiftB);
  };
  auto runConv = [&](int ci_, const float* act, const int* nb_, int N,
                     bool prelude, const float* res, float* out_) {
    const ConvP& c = convs[ci_];
    size_t sh = (size_t)ROWS * c.Kpad * 2;  // <= 294912 B, within 320KB WGP LDS
    if (prelude) {
      fconv_kernel<true><<<(N + ROWS - 1) / ROWS, THREADS, sh, stream>>>(
          act, nb_, c.pack, c.b, scaleB, shiftB, res, out_, N, c.cin, c.cout,
          c.Kpad);
    } else {
      fconv_kernel<false><<<(N + ROWS - 1) / ROWS, THREADS, sh, stream>>>(
          act, nb_, c.pack, c.b, nullptr, nullptr, res, out_, N, c.cin, c.cout,
          c.Kpad);
    }
  };
  auto runBlock = [&](int b, const int* nb_, int N, const float* in, float* t1,
                      float* out_) {
    runBn(in, N, binfo[b].cin, blk[b].n0);
    runConv(blk[b].c0, in, nb_, N, true, nullptr, t1);
    runBn(t1, N, binfo[b].cout, blk[b].n1);
    if (blk[b].ns >= 0) {
      runConv(blk[b].ns, in, nb_, N, false, nullptr, out_);  // shortcut
      runConv(blk[b].c1, t1, nb_, N, true, out_, out_);      // + residual
    } else {
      runConv(blk[b].c1, t1, nb_, N, true, in, out_);        // + identity
    }
  };

  float* A = buf[0];
  float* B = buf[1];
  float* T = buf[2];
  auto swapAB = [&]() { float* t = A; A = B; B = t; };

  // ---- Encoder ----
  runConv(encConvIn, x, nbh[0], NL[0], false, nullptr, A);
  int b = 0;
  for (int lvl = 0; lvl < 5; ++lvl) {
    runBlock(b++, nbh[lvl], NL[lvl], A, T, B); swapAB();
    runBlock(b++, nbh[lvl], NL[lvl], A, T, B); swapAB();
    int Cp = binfo[b - 1].cout;
    int Nout = NL[lvl + 1];
    pool4_kernel<<<((size_t)Nout * Cp + THREADS - 1) / THREADS, THREADS, 0,
                   stream>>>(A, B, Nout, Cp);
    swapAB();
  }
  runBlock(b++, nbh[5], NL[5], A, T, B); swapAB();  // enc_mid_0
  runBlock(b++, nbh[5], NL[5], A, T, B); swapAB();  // enc_mid_1

  // ---- Bottleneck ----
  runBn(A, NL[5], 256, encOutNorm);
  runConv(encOutConv, A, nbh[5], NL[5], true, nullptr, B); swapAB();
  runConv(decConvIn, A, nbh[5], NL[5], false, nullptr, B); swapAB();
  for (int k = 0; k < 5; ++k) {  // dec_mid x2 + up_4 x3 at level 5
    runBlock(b++, nbh[5], NL[5], A, T, B); swapAB();
  }

  // ---- Decoder ----
  for (int lvl = 4; lvl >= 1; --lvl) {
    int Cc = binfo[b].cin;  // channels entering this level
    int Nf = NL[lvl];
    blur_unpool_kernel<<<((size_t)Nf * Cc + THREADS - 1) / THREADS, THREADS, 0,
                         stream>>>(A, nbh[lvl], B, Nf, Cc);
    swapAB();
    runBlock(b++, nbh[lvl], Nf, A, T, B); swapAB();
    runBlock(b++, nbh[lvl], Nf, A, T, B); swapAB();
    runBlock(b++, nbh[lvl], Nf, A, T, B); swapAB();
  }
  {
    int Cc = binfo[29].cin;  // 64
    blur_unpool_kernel<<<((size_t)NL[0] * Cc + THREADS - 1) / THREADS, THREADS,
                         0, stream>>>(A, nbh[0], B, NL[0], Cc);
    swapAB();
    runBlock(29, nbh[0], NL[0], A, T, B); swapAB();  // dec_out_block
  }

  // ---- Head ----
  runBn(A, NL[0], 64, decOutNorm);
  head_kernel<<<(NL[0] + THREADS - 1) / THREADS, THREADS, 0, stream>>>(
      A, scaleB, shiftB, linW, linB, (float*)d_out, NL[0]);
}
